// GNN2_79783312490853
// MI455X (gfx1250) — matrix operand order repaired and checked by reading, not compiled, not run on hardware
//
#include <hip/hip_runtime.h>
#include <hip/hip_bf16.h>

#define N_DRUG  572
#define N_PAD   576          // padded to multiple of 16 for WMMA tiles
#define EMB     512
#define TWO_EMB 1024
#define K_NB    64
#define SEG_NB  16
#define BN_EPS  1e-5f

typedef __attribute__((ext_vector_type(16))) _Float16 v16h;
typedef __attribute__((ext_vector_type(8)))  _Float16 v8h;
typedef __attribute__((ext_vector_type(8)))  float    v8f;

static __device__ __forceinline__ v16h cat16(v8h lo, v8h hi) {
    return __builtin_shufflevector(lo, hi,
               0, 1, 2, 3, 4, 5, 6, 7, 8, 9, 10, 11, 12, 13, 14, 15);
}

// ---------------------------------------------------------------------------
// Kernel 1: convert + pack W_lin (f32 [1024x512]) into WMMA-fragment-major f16:
//   Wt[kb][tn][lane][j] = W[(kb*32 + (j>>3)*16 + (lane>>4)*8 + (j&7))*512 + tn*16 + (lane&15)]
// so each B fragment in the GEMM is two aligned b128 loads per lane.
// ---------------------------------------------------------------------------
__global__ __launch_bounds__(256) void k_pack_w(const float* __restrict__ W,
                                                _Float16* __restrict__ Wt) {
    const int i = blockIdx.x * blockDim.x + threadIdx.x;   // 0 .. 1024*512-1
    if (i >= TWO_EMB * EMB) return;
    const int j  = i & 15;          // half index within fragment
    const int L  = (i >> 4) & 31;   // lane
    const int tn = (i >> 9) & 31;   // n-tile
    const int kb = i >> 14;         // k-block (32 K per block)
    const int k  = kb * 32 + (j >> 3) * 16 + (L >> 4) * 8 + (j & 7);
    const int n  = tn * 16 + (L & 15);
    Wt[i] = (_Float16)W[(size_t)k * EMB + n];
}

// ---------------------------------------------------------------------------
// Kernel 2: fused gather + attention. One block per drug (576 blocks; the
// 4 pad blocks zero their e-row so the GEMM needs no M-edge guards).
//   scores_k = softmax_k( drug . rel_k );  attended = sum_k scores_k * ent_k
//   e[n] = concat(attended, drug_emb) in f16.
// Pure bandwidth stage (~150MB of embedding gathers -> ~6.5us roof @23.3TB/s).
// ---------------------------------------------------------------------------
__global__ __launch_bounds__(256) void k_attn(const int* __restrict__ drug_name,
                                              const int* __restrict__ adj_tail,
                                              const int* __restrict__ adj_rel,
                                              const float* __restrict__ drug_table,
                                              const float* __restrict__ rela_table,
                                              const float* __restrict__ ent_table,
                                              _Float16* __restrict__ e) {
    __shared__ float s_drug[EMB];
    __shared__ float s_sc[K_NB];

    const int n   = blockIdx.x;
    const int tid = threadIdx.x;

    if (n >= N_DRUG) {  // zero the pad rows of e
        for (int d = tid; d < TWO_EMB; d += 256)
            e[(size_t)n * TWO_EMB + d] = (_Float16)0.f;
        return;
    }

    const int dn = drug_name[n];
    for (int d = tid; d < EMB; d += 256)
        s_drug[d] = drug_table[(size_t)dn * EMB + d];
    __syncthreads();

    // 8 waves x 8 neighbors each: dot(drug, rel_k) with wave32 shuffle reduce
    const int wave = tid >> 5;
    const int lane = tid & 31;
    for (int kk = 0; kk < 8; ++kk) {
        const int k = wave * 8 + kk;
        const float* rrow = rela_table + (size_t)adj_rel[n * K_NB + k] * EMB;
        float p = 0.f;
        for (int j = lane; j < EMB; j += 32) p += s_drug[j] * rrow[j];
        #pragma unroll
        for (int off = 16; off > 0; off >>= 1) p += __shfl_down(p, off, 32);
        if (lane == 0) s_sc[k] = p;
    }
    __syncthreads();

    // tiny 64-way softmax: serial on thread 0 (negligible)
    if (tid == 0) {
        float mx = s_sc[0];
        for (int k = 1; k < K_NB; ++k) mx = fmaxf(mx, s_sc[k]);
        float s = 0.f;
        for (int k = 0; k < K_NB; ++k) { float ev = __expf(s_sc[k] - mx); s_sc[k] = ev; s += ev; }
        const float inv = 1.f / s;
        for (int k = 0; k < K_NB; ++k) s_sc[k] *= inv;
    }
    __syncthreads();

    // attended[d] = sum_k s_sc[k] * ent[tail_k][d]; coalesced across threads
    const int* tails = adj_tail + n * K_NB;
    for (int d = tid; d < EMB; d += 256) {
        float acc = 0.f;
        for (int k = 0; k < K_NB; ++k)
            acc += s_sc[k] * ent_table[(size_t)tails[k] * EMB + d];
        e[(size_t)n * TWO_EMB + d]       = (_Float16)acc;
        e[(size_t)n * TWO_EMB + EMB + d] = (_Float16)s_drug[d];
    }
}

// ---------------------------------------------------------------------------
// Kernel 3: h = relu(e @ W + b) via v_wmma_f32_16x16x32_f16.
// One wave computes a 16x64 strip: 4 N-tiles reuse one A fragment per k-step.
// Ping-pong software pipeline (unroll-by-2): while set0's 4 wmmas execute,
// set1's 10 b128 loads are in flight, and vice versa. Each fragment is
// defined once and consumed immediately by its wmma -> no register rotation.
// Fragment layouts per CDNA5 ISA 7.12.2 (wave32):
//   A 16x32 f16 : lane m = lane&15; halfs [k0+half*8, +7] and [k0+16+half*8, +7]
//   B 32x16 f16 : pre-packed fragment-major by k_pack_w
//   C/D 16x16 f32: VGPR r -> M = r + 8*(lane>>4), N = lane&15
// ---------------------------------------------------------------------------
__global__ __launch_bounds__(128) void k_gemm(const _Float16* __restrict__ e,   // [N_PAD][1024]
                                              const _Float16* __restrict__ Wt,  // packed [32][32][32][16]
                                              const float* __restrict__ bias,   // [512]
                                              float* __restrict__ h) {          // [N_PAD][512]
    const int lane = threadIdx.x & 31;
    const int wid  = blockIdx.x * 4 + (threadIdx.x >> 5);  // 0..287
    const int tm   = wid >> 3;                             // 0..35
    const int ng   = wid & 7;                              // n-group (4 tiles)
    const int m0   = tm * 16;
    const int half = lane >> 4;
    const int l15  = lane & 15;

    // per-lane base pointers
    const _Float16* arow  = e + (size_t)(m0 + l15) * TWO_EMB + half * 8;
    // Wt offset for (kb, tn, lane): kb*16384 + tn*512 + lane*16 halfs
    const _Float16* bbase = Wt + (size_t)(ng * 4) * 512 + (size_t)lane * 16;

    v8f c[4] = {v8f{}, v8f{}, v8f{}, v8f{}};

    // preload set0 (kb = 0)
    v8h al0 = *(const v8h*)(arow);
    v8h ah0 = *(const v8h*)(arow + 16);
    v8h bl0_0 = *(const v8h*)(bbase);          v8h bh0_0 = *(const v8h*)(bbase + 8);
    v8h bl0_1 = *(const v8h*)(bbase + 512);    v8h bh0_1 = *(const v8h*)(bbase + 520);
    v8h bl0_2 = *(const v8h*)(bbase + 1024);   v8h bh0_2 = *(const v8h*)(bbase + 1032);
    v8h bl0_3 = *(const v8h*)(bbase + 1536);   v8h bh0_3 = *(const v8h*)(bbase + 1544);

    for (int kb = 0; kb < 32; kb += 2) {
        // ---- issue set1 loads (kb+1); independent of set0's wmmas ----
        const _Float16* a1 = arow  + (size_t)(kb + 1) * 32;
        const _Float16* b1 = bbase + (size_t)(kb + 1) * 16384;
        v8h al1 = *(const v8h*)(a1);
        v8h ah1 = *(const v8h*)(a1 + 16);
        v8h bl1_0 = *(const v8h*)(b1);          v8h bh1_0 = *(const v8h*)(b1 + 8);
        v8h bl1_1 = *(const v8h*)(b1 + 512);    v8h bh1_1 = *(const v8h*)(b1 + 520);
        v8h bl1_2 = *(const v8h*)(b1 + 1024);   v8h bh1_2 = *(const v8h*)(b1 + 1032);
        v8h bl1_3 = *(const v8h*)(b1 + 1536);   v8h bh1_3 = *(const v8h*)(b1 + 1544);

        // ---- consume set0 ----
        {
            const v16h a = cat16(al0, ah0);
            c[0] = __builtin_amdgcn_wmma_f32_16x16x32_f16(false, a, false, cat16(bl0_0, bh0_0),
                                                          (short)0, c[0], false, false);
            c[1] = __builtin_amdgcn_wmma_f32_16x16x32_f16(false, a, false, cat16(bl0_1, bh0_1),
                                                          (short)0, c[1], false, false);
            c[2] = __builtin_amdgcn_wmma_f32_16x16x32_f16(false, a, false, cat16(bl0_2, bh0_2),
                                                          (short)0, c[2], false, false);
            c[3] = __builtin_amdgcn_wmma_f32_16x16x32_f16(false, a, false, cat16(bl0_3, bh0_3),
                                                          (short)0, c[3], false, false);
        }

        // ---- issue set0 loads (kb+2); overlap with set1's wmmas ----
        if (kb + 2 < 32) {
            const _Float16* a2 = arow  + (size_t)(kb + 2) * 32;
            const _Float16* b2 = bbase + (size_t)(kb + 2) * 16384;
            al0 = *(const v8h*)(a2);
            ah0 = *(const v8h*)(a2 + 16);
            bl0_0 = *(const v8h*)(b2);          bh0_0 = *(const v8h*)(b2 + 8);
            bl0_1 = *(const v8h*)(b2 + 512);    bh0_1 = *(const v8h*)(b2 + 520);
            bl0_2 = *(const v8h*)(b2 + 1024);   bh0_2 = *(const v8h*)(b2 + 1032);
            bl0_3 = *(const v8h*)(b2 + 1536);   bh0_3 = *(const v8h*)(b2 + 1544);
        }

        // ---- consume set1 ----
        {
            const v16h a = cat16(al1, ah1);
            c[0] = __builtin_amdgcn_wmma_f32_16x16x32_f16(false, a, false, cat16(bl1_0, bh1_0),
                                                          (short)0, c[0], false, false);
            c[1] = __builtin_amdgcn_wmma_f32_16x16x32_f16(false, a, false, cat16(bl1_1, bh1_1),
                                                          (short)0, c[1], false, false);
            c[2] = __builtin_amdgcn_wmma_f32_16x16x32_f16(false, a, false, cat16(bl1_2, bh1_2),
                                                          (short)0, c[2], false, false);
            c[3] = __builtin_amdgcn_wmma_f32_16x16x32_f16(false, a, false, cat16(bl1_3, bh1_3),
                                                          (short)0, c[3], false, false);
        }
    }

    #pragma unroll
    for (int t = 0; t < 4; ++t) {
        const int n  = (ng * 4 + t) * 16 + l15;
        const float bv = bias[n];
        #pragma unroll
        for (int r = 0; r < 8; ++r) {
            const int m = m0 + r + half * 8;
            h[(size_t)m * EMB + n] = fmaxf(c[t][r] + bv, 0.f);
        }
    }
}

// ---------------------------------------------------------------------------
// Kernel 4: BatchNorm1d (training stats, biased var). One 512-thread block,
// thread d owns feature column d; one-pass sum/sumsq then normalize.
// ---------------------------------------------------------------------------
__global__ __launch_bounds__(512) void k_bn(const float* __restrict__ h,
                                            const float* __restrict__ gamma,
                                            const float* __restrict__ beta,
                                            float* __restrict__ out) {
    const int d = threadIdx.x;
    float s = 0.f, s2 = 0.f;
    for (int n = 0; n < N_DRUG; ++n) {
        const float v = h[(size_t)n * EMB + d];
        s += v; s2 += v * v;
    }
    const float mean  = s * (1.f / N_DRUG);
    const float var   = s2 * (1.f / N_DRUG) - mean * mean;
    const float scale = gamma[d] * rsqrtf(var + BN_EPS);
    const float shift = beta[d] - mean * scale;
    for (int n = 0; n < N_DRUG; ++n)
        out[(size_t)n * EMB + d] = h[(size_t)n * EMB + d] * scale + shift;
}

// ---------------------------------------------------------------------------
// Kernel 5: sequential neighbor averaging (later rows see updated rows).
// Feature columns are independent: thread d only touches column d, and the
// neighbor indices are wave-uniform (scalarized loads) -> zero barriers.
// Same-lane store->load to the same address is hardware-ordered (ISA 7.3).
// ---------------------------------------------------------------------------
__global__ __launch_bounds__(512) void k_scan(const int* __restrict__ inv_idx,
                                              const int* __restrict__ epoch,
                                              float* __restrict__ out) {
    if (*epoch <= 1) return;
    const int d = threadIdx.x;
    for (int i = 0; i < N_DRUG; ++i) {
        const int* iv = inv_idx + i * SEG_NB;
        float acc = 0.f;
        #pragma unroll
        for (int j = 0; j < SEG_NB; ++j)
            acc += out[(size_t)iv[j] * EMB + d];
        const float cur = out[(size_t)i * EMB + d];
        out[(size_t)i * EMB + d] = 0.5f * (acc * (1.f / SEG_NB) + cur);
    }
}

// ---------------------------------------------------------------------------
extern "C" void kernel_launch(void* const* d_in, const int* in_sizes, int n_in,
                              void* d_out, int out_size, void* d_ws, size_t ws_size,
                              hipStream_t stream) {
    const int*   drug_name  = (const int*)d_in[0];
    const int*   adj_tail   = (const int*)d_in[1];
    const int*   adj_rel    = (const int*)d_in[2];
    const int*   inv_idx    = (const int*)d_in[3];
    const float* drug_table = (const float*)d_in[4];
    const float* rela_table = (const float*)d_in[5];
    const float* ent_table  = (const float*)d_in[6];
    const float* W          = (const float*)d_in[7];
    const float* b          = (const float*)d_in[8];
    const float* gamma      = (const float*)d_in[9];
    const float* beta       = (const float*)d_in[10];
    const int*   epoch      = (const int*)d_in[11];
    float*       out        = (float*)d_out;

    // workspace: e_f16 [576x1024] | Wt_f16 packed [1024x512] | h_f32 [576x512]
    char* ws = (char*)d_ws;
    _Float16* e  = (_Float16*)ws;
    _Float16* Wt = (_Float16*)(ws + (size_t)N_PAD * TWO_EMB * sizeof(_Float16));
    float*    h  = (float*)(ws + (size_t)N_PAD * TWO_EMB * sizeof(_Float16)
                               + (size_t)TWO_EMB * EMB * sizeof(_Float16));

    k_pack_w<<<(TWO_EMB * EMB + 255) / 256, 256, 0, stream>>>(W, Wt);
    k_attn<<<N_PAD, 256, 0, stream>>>(drug_name, adj_tail, adj_rel,
                                      drug_table, rela_table, ent_table, e);
    k_gemm<<<(N_PAD / 16) * 8 / 4, 128, 0, stream>>>(e, Wt, b, h);
    k_bn<<<1, 512, 0, stream>>>(h, gamma, beta, out);
    k_scan<<<1, 512, 0, stream>>>(inv_idx, epoch, out);
}
